// GenericMoeDecoderLayer_5952824672538
// MI455X (gfx1250) — matrix operand order repaired and checked by reading, not compiled, not run on hardware
//
#include <hip/hip_runtime.h>
#include <hip/hip_bf16.h>

// ---------- CDNA5 WMMA types ----------
typedef __attribute__((ext_vector_type(16))) __bf16 v16bf;
typedef __attribute__((ext_vector_type(8)))  __bf16 v8bf;
typedef __attribute__((ext_vector_type(8)))  float  v8f;
typedef __attribute__((ext_vector_type(4)))  unsigned int u32x4;
typedef __attribute__((ext_vector_type(8)))  int i32x8;
typedef __attribute__((ext_vector_type(4)))  int i32x4;

union FragU { v16bf v; v8bf h[2]; };

// ---------- problem constants ----------
#define TT 2048
#define HH 2048
#define EE 64
#define KK 8
#define II 768
#define NHH 16
#define DHH 128
#define SII 4096
#define CAPP 512

// ---------- GEMM tile config ----------
#define BM 128
#define BN 128
#define BK 32
#define LDT 40   // LDS row stride in bf16 (32 + 8 pad) = 80 bytes = 16 DW + 4 DW pad

// TDM: load a 128-row x 32-elem bf16 tile (row stride K elems) into LDS with
// the 16DW-data/4DW-pad layout matching LDT. Issued wave-uniform.
__device__ __forceinline__
void tdm_load_tile_bf16(unsigned lds_off, const __bf16* gptr, int K, int M)
{
    unsigned long long ga = (unsigned long long)(const void*)gptr;
    u32x4 g0;
    g0[0] = 1u;                                    // count=1, user mode
    g0[1] = lds_off;                               // lds_addr [63:32]
    g0[2] = (unsigned)(ga & 0xffffffffull);        // global_addr [95:64]
    g0[3] = (unsigned)((ga >> 32) & 0x1ffffffull)  // global_addr [120:96]
          | (2u << 30);                            // type=2 (image) [127:126]
    i32x8 g1;
    g1[0] = (1 << 16)       // data_size = 2 bytes
          | (1 << 20)       // pad_enable
          | (3 << 22)       // pad_interval: code 3 = 16 DWORDs of data
          | (3 << 25);      // pad_amount:   code 3 = 4 DWORDs of pad
    g1[1] = (int)(((unsigned)K & 0xffffu) << 16);              // tensor_dim0 lo16 [63:48]
    g1[2] = (int)((((unsigned)K) >> 16) | (((unsigned)M & 0xffffu) << 16)); // dim0 hi / dim1 lo
    g1[3] = (int)((((unsigned)M) >> 16) | (32u << 16));        // dim1 hi | tile_dim0=32
    g1[4] = 128;                                               // tile_dim1=128, tile_dim2=0
    g1[5] = K;                                                 // tensor_dim0_stride lo32
    g1[6] = 0;
    g1[7] = 0;
    i32x4 gz = {0, 0, 0, 0};
#if defined(__clang_major__) && (__clang_major__ >= 23)
    i32x8 gz8 = {0, 0, 0, 0, 0, 0, 0, 0};
    __builtin_amdgcn_tensor_load_to_lds(g0, g1, gz, gz, gz8, 0);
#else
    __builtin_amdgcn_tensor_load_to_lds(g0, g1, gz, gz, 0);
#endif
}

// C[M,N](f32 acc) = A[M,K]_bf16 * op(B_f32)   op = B^T if B_NK (B stored [N,K]) else B ([K,N])
// optional: += addsrc (f32), output f32 or bf16. batched via blockIdx.z with strides.
template<bool B_NK, bool OUT_BF16, bool ADD_IN>
__global__ __launch_bounds__(256)
void gemm_wmma(const __bf16* __restrict__ A, const float* __restrict__ B,
               void* __restrict__ C, const float* __restrict__ addsrc,
               int M, int N, int K,
               long strideA, long strideB, long strideC)
{
    __shared__ __bf16 As[BM * LDT];
    __shared__ __bf16 Bs[BN * LDT];

    const int tid  = threadIdx.x;
    const int lane = tid & 31;
    const int w    = tid >> 5;        // 0..7
    const int wm   = w & 1;           // 2 -> 64 rows each
    const int wn   = w >> 1;          // 4 -> 32 cols each
    const int half = lane >> 4;
    const int l15  = lane & 15;

    const int m0 = blockIdx.y * BM;
    const int n0 = blockIdx.x * BN;

    const __bf16* Ab = A + (long)blockIdx.z * strideA;
    const float*  Bb = B + (long)blockIdx.z * strideB;

    const unsigned as_off = (unsigned)(unsigned long long)(const void*)&As[0];

    v8f acc[4][2];
#pragma unroll
    for (int mt = 0; mt < 4; ++mt)
#pragma unroll
        for (int nt = 0; nt < 2; ++nt)
            acc[mt][nt] = {};

    for (int k0 = 0; k0 < K; k0 += BK) {
        __syncthreads();
        // ---- A tile via Tensor Data Mover (bf16, padded rows land directly in LDT layout)
        if (w == 0) {
            tdm_load_tile_bf16(as_off, Ab + (long)m0 * K + k0, K, M);
            __builtin_amdgcn_s_wait_tensorcnt(0);
        }
        // ---- stage B tile (f32 -> bf16 LDS as [n][k]) ----
        if (B_NK) {
#pragma unroll
            for (int i = 0; i < 4; ++i) {
                int lin = tid + i * 256;        // 1024 float4 chunks
                int row = lin >> 3, c4 = lin & 7;
                int gn = n0 + row;
                float4 v = make_float4(0.f, 0.f, 0.f, 0.f);
                if (gn < N) v = *(const float4*)(Bb + (long)gn * K + k0 + c4 * 4);
                __bf16* d = &Bs[row * LDT + c4 * 4];
                d[0] = (__bf16)v.x; d[1] = (__bf16)v.y;
                d[2] = (__bf16)v.z; d[3] = (__bf16)v.w;
            }
        } else {
#pragma unroll
            for (int i = 0; i < 16; ++i) {
                int lin = tid + i * 256;        // 4096 scalars
                int n = lin & 127, kk = lin >> 7;
                int gn = n0 + n;
                float v = 0.f;
                if (gn < N) v = Bb[(long)(k0 + kk) * N + gn];
                Bs[n * LDT + kk] = (__bf16)v;
            }
        }
        __syncthreads();

        // prefetch next B tile (global_prefetch_b8, speculative)
        if (k0 + BK < K) {
            const float* pf = B_NK ? (Bb + (long)(n0 + (tid & 127)) * K + k0 + BK)
                                   : (Bb + (long)(k0 + BK + (tid >> 3)) * N + n0);
            __builtin_prefetch((const void*)pf, 0, 1);
        }

        // ---- fragments ----
        v16bf af[4];
#pragma unroll
        for (int mt = 0; mt < 4; ++mt) {
            const __bf16* p = &As[(wm * 64 + mt * 16 + l15) * LDT + half * 8];
            FragU u;
            u.h[0] = *(const v8bf*)p;           // K kbase .. kbase+7
            u.h[1] = *(const v8bf*)(p + 16);    // K kbase+16 .. kbase+23
            af[mt] = u.v;
        }
        v16bf bfr[2];
#pragma unroll
        for (int nt = 0; nt < 2; ++nt) {
            const __bf16* p = &Bs[(wn * 32 + nt * 16 + l15) * LDT + half * 8];
            FragU u;
            u.h[0] = *(const v8bf*)p;           // K kb .. kb+7
            u.h[1] = *(const v8bf*)(p + 16);    // K kb+16 .. kb+23
            bfr[nt] = u.v;
        }
#pragma unroll
        for (int mt = 0; mt < 4; ++mt)
#pragma unroll
            for (int nt = 0; nt < 2; ++nt)
                acc[mt][nt] = __builtin_amdgcn_wmma_f32_16x16x32_bf16(
                    false, af[mt], false, bfr[nt], (short)0, acc[mt][nt], false, false);
    }

    // ---- store C ----
#pragma unroll
    for (int mt = 0; mt < 4; ++mt) {
#pragma unroll
        for (int nt = 0; nt < 2; ++nt) {
            int gm0 = m0 + wm * 64 + mt * 16 + half * 8;
            int gn  = n0 + wn * 32 + nt * 16 + l15;
            if (gn >= N) continue;
#pragma unroll
            for (int r = 0; r < 8; ++r) {
                int gm = gm0 + r;
                if (gm >= M) continue;
                long idx = (long)blockIdx.z * strideC + (long)gm * N + gn;
                float v = acc[mt][nt][r];
                if (ADD_IN) v += addsrc[idx];
                if (OUT_BF16) ((__bf16*)C)[idx] = (__bf16)v;
                else          ((float*)C)[idx]  = v;
            }
        }
    }
}

// ---------- RMSNorm (f32 in, bf16 out) ----------
__global__ __launch_bounds__(256)
void rmsnorm_kernel(const float* __restrict__ x, const float* __restrict__ g,
                    __bf16* __restrict__ out)
{
    const int row  = blockIdx.x;
    const int tid  = threadIdx.x;
    const int lane = tid & 31;
    const int w    = tid >> 5;
    const float* xr = x + (long)row * HH;

    float s = 0.f;
#pragma unroll
    for (int i = 0; i < HH / 256; ++i) { float v = xr[tid + i * 256]; s += v * v; }
#pragma unroll
    for (int m = 16; m >= 1; m >>= 1) s += __shfl_xor(s, m, 32);

    __shared__ float red[8];
    if (lane == 0) red[w] = s;
    __syncthreads();
    float tot = 0.f;
#pragma unroll
    for (int i = 0; i < 8; ++i) tot += red[i];
    float rinv = rsqrtf(tot / (float)HH + 1e-6f);

#pragma unroll
    for (int i = 0; i < HH / 256; ++i) {
        int c = tid + i * 256;
        out[(long)row * HH + c] = (__bf16)(xr[c] * rinv * g[c]);
    }
}

// ---------- Flash attention: one (head, 128-row q block) per CTA ----------
__global__ __launch_bounds__(256)
void attn_kernel(const __bf16* __restrict__ qkv, __bf16* __restrict__ attn_o)
{
    const int head = blockIdx.x;   // 16
    const int qb   = blockIdx.y;   // 16
    const int tid  = threadIdx.x;
    const int lane = tid & 31;
    const int w    = tid >> 5;     // wave: 16 q rows each
    const int half = lane >> 4;
    const int l15  = lane & 15;

    __shared__ __bf16 Ks[32 * 136];      // [kv tok][dh], stride 136 (272B)
    __shared__ __bf16 Vt[128 * 40];      // [dh][kv tok], stride 40 (80B)
    __shared__ __bf16 Ps[8 * 16 * 40];   // per-wave P tile 16x32, stride 40

    // Q fragments straight from global (A-frag layout)
    v16bf qf[4];
    {
        int qrow = qb * 128 + w * 16 + l15;
        const __bf16* qp = qkv + (long)qrow * (3 * HH) + head * DHH;
#pragma unroll
        for (int kc = 0; kc < 4; ++kc) {
            const __bf16* p = qp + kc * 32 + half * 8;
            FragU u;
            u.h[0] = *(const v8bf*)p;
            u.h[1] = *(const v8bf*)(p + 16);
            qf[kc] = u.v;
        }
    }

    v8f o[8];
#pragma unroll
    for (int nt = 0; nt < 8; ++nt) o[nt] = {};
    float mi[8], li[8];
#pragma unroll
    for (int r = 0; r < 8; ++r) { mi[r] = -1e30f; li[r] = 0.f; }

    const float sc = 0.08838834764831845f;   // 1/sqrt(128)
    const int jmax = (qb + 1) * 4;           // 32-token kv tiles (causal)

    for (int j = 0; j < jmax; ++j) {
        const int kv0 = j * 32;
        __syncthreads();
        // K tile -> LDS  (rows = kv token)
#pragma unroll
        for (int i = 0; i < 2; ++i) {
            int lin = tid + i * 256;         // 512 chunks of 8
            int row = lin >> 4, c8 = lin & 15;
            v8bf v = *(const v8bf*)(qkv + (long)(kv0 + row) * (3 * HH) + HH + head * DHH + c8 * 8);
            *(v8bf*)&Ks[row * 136 + c8 * 8] = v;
        }
        // V tile transposed -> LDS [dh][tok]
#pragma unroll
        for (int i = 0; i < 16; ++i) {
            int lin = tid + i * 256;         // 4096 scalars
            int tok = lin >> 7, dh = lin & 127;
            Vt[dh * 40 + tok] = qkv[(long)(kv0 + tok) * (3 * HH) + 2 * HH + head * DHH + dh];
        }
        __syncthreads();

        // S = Q K^T  (two 16x16 tiles over K=128)
        v8f S0 = {}, S1 = {};
#pragma unroll
        for (int kc = 0; kc < 4; ++kc) {
            FragU u0, u1;
            const __bf16* p0 = &Ks[(0  + l15) * 136 + kc * 32 + half * 8];
            const __bf16* p1 = &Ks[(16 + l15) * 136 + kc * 32 + half * 8];
            u0.h[0] = *(const v8bf*)p0;  u0.h[1] = *(const v8bf*)(p0 + 16);
            u1.h[0] = *(const v8bf*)p1;  u1.h[1] = *(const v8bf*)(p1 + 16);
            S0 = __builtin_amdgcn_wmma_f32_16x16x32_bf16(false, qf[kc], false, u0.v, (short)0, S0, false, false);
            S1 = __builtin_amdgcn_wmma_f32_16x16x32_bf16(false, qf[kc], false, u1.v, (short)0, S1, false, false);
        }

        // online softmax (fully unrolled: accumulator element indices must be static)
        const int rowbase = qb * 128 + w * 16 + half * 8;
        float p0v[8], p1v[8];
#pragma unroll
        for (int r = 0; r < 8; ++r) {
            int row  = rowbase + r;
            int col0 = kv0 + l15;
            int col1 = kv0 + 16 + l15;
            float s0 = S0[r] * sc, s1 = S1[r] * sc;
            bool msk0 = (col0 > row), msk1 = (col1 > row);
            if (msk0) s0 = -1e30f;
            if (msk1) s1 = -1e30f;
            float m = fmaxf(s0, s1);
#pragma unroll
            for (int msk = 8; msk >= 1; msk >>= 1) m = fmaxf(m, __shfl_xor(m, msk, 32));
            float mn = fmaxf(mi[r], m);
            float scale = __expf(mi[r] - mn);
            float e0 = msk0 ? 0.f : __expf(s0 - mn);
            float e1 = msk1 ? 0.f : __expf(s1 - mn);
            float sum = e0 + e1;
#pragma unroll
            for (int msk = 8; msk >= 1; msk >>= 1) sum += __shfl_xor(sum, msk, 32);
            li[r] = li[r] * scale + sum;
            mi[r] = mn;
            p0v[r] = e0; p1v[r] = e1;
#pragma unroll
            for (int nt = 0; nt < 8; ++nt) o[nt][r] *= scale;
        }

        // P (C layout) -> per-wave LDS -> A-frag
        __bf16* ps = &Ps[w * 16 * 40];
#pragma unroll
        for (int r = 0; r < 8; ++r) {
            ps[(half * 8 + r) * 40 + l15]      = (__bf16)p0v[r];
            ps[(half * 8 + r) * 40 + 16 + l15] = (__bf16)p1v[r];
        }
        FragU up;
        {
            const __bf16* p = &ps[l15 * 40 + half * 8];
            up.h[0] = *(const v8bf*)p;
            up.h[1] = *(const v8bf*)(p + 16);
        }
        // O += P V   (8 N-tiles over DH=128)
#pragma unroll
        for (int nt = 0; nt < 8; ++nt) {
            FragU ub;
            const __bf16* p = &Vt[(nt * 16 + l15) * 40 + half * 8];
            ub.h[0] = *(const v8bf*)p;  ub.h[1] = *(const v8bf*)(p + 16);
            o[nt] = __builtin_amdgcn_wmma_f32_16x16x32_bf16(false, up.v, false, ub.v, (short)0, o[nt], false, false);
        }
    }

    // normalize + store
    float inv[8];
#pragma unroll
    for (int r = 0; r < 8; ++r) inv[r] = 1.f / li[r];
    const int t0 = qb * 128 + w * 16 + half * 8;
#pragma unroll
    for (int nt = 0; nt < 8; ++nt) {
        int dh = head * DHH + nt * 16 + l15;
#pragma unroll
        for (int r = 0; r < 8; ++r)
            attn_o[(long)(t0 + r) * HH + dh] = (__bf16)(o[nt][r] * inv[r]);
    }
}

// ---------- Routing: softmax + top-8 per token (deterministic) ----------
__global__ __launch_bounds__(256)
void route_topk(const float* __restrict__ logits, int* __restrict__ flat_e,
                float* __restrict__ flat_w)
{
    int t = blockIdx.x * 256 + threadIdx.x;
    if (t >= TT) return;
    const float* l = logits + (long)t * EE;
    float mx = -1e30f;
    for (int e = 0; e < EE; ++e) mx = fmaxf(mx, l[e]);
    float p[EE]; float sum = 0.f;
    for (int e = 0; e < EE; ++e) { p[e] = __expf(l[e] - mx); sum += p[e]; }
    float inv = 1.f / sum;
    unsigned long long used = 0ull;
    float tot = 0.f; int ids[KK]; float vals[KK];
    for (int k = 0; k < KK; ++k) {
        float best = -1.f; int bi = 0;
        for (int e = 0; e < EE; ++e)
            if (!((used >> e) & 1ull) && p[e] > best) { best = p[e]; bi = e; }
        used |= (1ull << bi);
        ids[k] = bi; vals[k] = best * inv; tot += best * inv;
    }
    float winv = 1.f / tot;
    for (int k = 0; k < KK; ++k) {
        flat_e[t * KK + k] = ids[k];
        flat_w[t * KK + k] = vals[k] * winv;
    }
}

// one thread per expert scans entries in token order -> matches reference cumsum positions
__global__ void route_pos(const int* __restrict__ flat_e, int* __restrict__ pos,
                          int* __restrict__ keep)
{
    int e = threadIdx.x;
    if (e >= EE) return;
    int cnt = 0;
    for (int i = 0; i < TT * KK; ++i) {
        if (flat_e[i] == e) {
            pos[i]  = cnt;
            keep[i] = (cnt < CAPP) ? 1 : 0;
            ++cnt;
        }
    }
}

// scatter kept tokens into capacity buffers
__global__ __launch_bounds__(256)
void moe_scatter(const int* __restrict__ flat_e, const int* __restrict__ pos,
                 const int* __restrict__ keep, const __bf16* __restrict__ h2,
                 __bf16* __restrict__ buf)
{
    int i = blockIdx.x;                 // entry id, T*K
    if (!keep[i]) return;
    int e = flat_e[i], p = pos[i], tok = i >> 3;
    const v8bf* src = (const v8bf*)(h2 + (long)tok * HH);
    v8bf* dst = (v8bf*)(buf + ((long)e * CAPP + p) * HH);
    dst[threadIdx.x] = src[threadIdx.x];   // 256 * 8 bf16 = 2048
}

// act = silu(g) * u over rows of [2*I] gu buffers
__global__ __launch_bounds__(256)
void silu_mul(const __bf16* __restrict__ gu, __bf16* __restrict__ act,
              long rows, int Iw)
{
    long idx = (long)blockIdx.x * 256 + threadIdx.x;
    if (idx >= rows * (long)Iw) return;
    long r = idx / Iw; int c = (int)(idx - r * Iw);
    float g = (float)gu[r * (2L * Iw) + c];
    float u = (float)gu[r * (2L * Iw) + Iw + c];
    float s = g / (1.f + __expf(-g));
    act[idx] = (__bf16)(s * u);
}

// out = x1 + shared_y + sum_k w_k * y_moe[e_k][pos_k]   (deterministic gather per token)
__global__ __launch_bounds__(256)
void final_combine(const float* __restrict__ x1, const float* __restrict__ sh_y,
                   const __bf16* __restrict__ y_moe,
                   const int* __restrict__ flat_e, const int* __restrict__ pos,
                   const int* __restrict__ keep, const float* __restrict__ flat_w,
                   float* __restrict__ out)
{
    int t = blockIdx.x;
    int es[KK], ps[KK], kp[KK]; float ww[KK];
#pragma unroll
    for (int k = 0; k < KK; ++k) {
        int idx = t * KK + k;
        es[k] = flat_e[idx]; ps[k] = pos[idx]; kp[k] = keep[idx]; ww[k] = flat_w[idx];
    }
#pragma unroll
    for (int i = 0; i < 8; ++i) {
        int col = threadIdx.x + i * 256;
        long bi = (long)t * HH + col;
        float acc = x1[bi] + sh_y[bi];
#pragma unroll
        for (int k = 0; k < KK; ++k)
            if (kp[k])
                acc += ww[k] * (float)y_moe[((long)es[k] * CAPP + ps[k]) * HH + col];
        out[bi] = acc;
    }
}

// ------------------------------------------------------------------
extern "C" void kernel_launch(void* const* d_in, const int* in_sizes, int n_in,
                              void* d_out, int out_size, void* d_ws, size_t ws_size,
                              hipStream_t stream)
{
    (void)in_sizes; (void)n_in; (void)out_size; (void)ws_size;
    const float* x        = (const float*)d_in[0];   // [T,H]
    const float* w_qkv    = (const float*)d_in[1];   // [H,3H]
    const float* w_o      = (const float*)d_in[2];   // [H,H]
    const float* pre_g    = (const float*)d_in[3];   // [H]
    const float* post_g   = (const float*)d_in[4];   // [H]
    const float* w_gate   = (const float*)d_in[5];   // [E,H]
    const float* w1       = (const float*)d_in[6];   // [E,2I,H]
    const float* w2       = (const float*)d_in[7];   // [E,H,I]
    const float* w13s     = (const float*)d_in[8];   // [2SI,H]
    const float* w2s      = (const float*)d_in[9];   // [H,SI]
    float* out = (float*)d_out;

    // ---- workspace carving (bytes) ----
    char* ws = (char*)d_ws;
    size_t off = 0;
    auto alloc = [&](size_t bytes) { char* p = ws + off; off += (bytes + 255) & ~(size_t)255; return p; };
    __bf16* h_bf    = (__bf16*)alloc((size_t)TT * HH * 2);
    __bf16* qkv_bf  = (__bf16*)alloc((size_t)TT * 3 * HH * 2);
    __bf16* ao_bf   = (__bf16*)alloc((size_t)TT * HH * 2);
    float*  x1      = (float*) alloc((size_t)TT * HH * 4);
    __bf16* h2_bf   = (__bf16*)alloc((size_t)TT * HH * 2);
    float*  logits  = (float*) alloc((size_t)TT * EE * 4);
    int*    flat_e  = (int*)   alloc((size_t)TT * KK * 4);
    float*  flat_w  = (float*) alloc((size_t)TT * KK * 4);
    int*    pos     = (int*)   alloc((size_t)TT * KK * 4);
    int*    keep    = (int*)   alloc((size_t)TT * KK * 4);
    __bf16* buf_bf  = (__bf16*)alloc((size_t)EE * CAPP * HH * 2);
    __bf16* gu_moe  = (__bf16*)alloc((size_t)EE * CAPP * 2 * II * 2);
    __bf16* act_moe = (__bf16*)alloc((size_t)EE * CAPP * II * 2);
    __bf16* y_moe   = (__bf16*)alloc((size_t)EE * CAPP * HH * 2);
    __bf16* gu_sh   = (__bf16*)alloc((size_t)TT * 2 * SII * 2);
    __bf16* act_sh  = (__bf16*)alloc((size_t)TT * SII * 2);
    float*  sh_y    = (float*) alloc((size_t)TT * HH * 4);

    // 1) pre-LN
    rmsnorm_kernel<<<TT, 256, 0, stream>>>(x, pre_g, h_bf);

    // 2) QKV = h @ w_qkv   (B is [K,N])
    gemm_wmma<false, true, false><<<dim3(3 * HH / BN, TT / BM, 1), 256, 0, stream>>>(
        h_bf, w_qkv, qkv_bf, nullptr, TT, 3 * HH, HH, 0, 0, 0);

    // 3) attention
    attn_kernel<<<dim3(NHH, TT / 128), 256, 0, stream>>>(qkv_bf, ao_bf);

    // 4) x1 = x + attn_o @ w_o
    gemm_wmma<false, false, true><<<dim3(HH / BN, TT / BM, 1), 256, 0, stream>>>(
        ao_bf, w_o, x1, x, TT, HH, HH, 0, 0, 0);

    // 5) post-LN
    rmsnorm_kernel<<<TT, 256, 0, stream>>>(x1, post_g, h2_bf);

    // 6) gate logits = h2 @ w_gate^T  (N=64)
    gemm_wmma<true, false, false><<<dim3(1, TT / BM, 1), 256, 0, stream>>>(
        h2_bf, w_gate, logits, nullptr, TT, EE, HH, 0, 0, 0);

    // 7-9) routing + scatter
    route_topk<<<TT / 256, 256, 0, stream>>>(logits, flat_e, flat_w);
    route_pos<<<1, 64, 0, stream>>>(flat_e, pos, keep);
    moe_scatter<<<TT * KK, 256, 0, stream>>>(flat_e, pos, keep, h2_bf, buf_bf);

    // 10) expert gu = buf @ w1^T   (batched over 64 experts)
    gemm_wmma<true, true, false><<<dim3(2 * II / BN, CAPP / BM, EE), 256, 0, stream>>>(
        buf_bf, w1, gu_moe, nullptr, CAPP, 2 * II, HH,
        (long)CAPP * HH, (long)2 * II * HH, (long)CAPP * 2 * II);

    // 11) act = silu(g)*u
    {
        long n = (long)EE * CAPP * II;
        silu_mul<<<(unsigned)((n + 255) / 256), 256, 0, stream>>>(gu_moe, act_moe, (long)EE * CAPP, II);
    }

    // 12) y = act @ w2^T
    gemm_wmma<true, true, false><<<dim3(HH / BN, CAPP / BM, EE), 256, 0, stream>>>(
        act_moe, w2, y_moe, nullptr, CAPP, HH, II,
        (long)CAPP * II, (long)HH * II, (long)CAPP * HH);

    // 13) shared gu = h2 @ w13^T
    gemm_wmma<true, true, false><<<dim3(2 * SII / BN, TT / BM, 1), 256, 0, stream>>>(
        h2_bf, w13s, gu_sh, nullptr, TT, 2 * SII, HH, 0, 0, 0);

    // 14) shared act
    {
        long n = (long)TT * SII;
        silu_mul<<<(unsigned)((n + 255) / 256), 256, 0, stream>>>(gu_sh, act_sh, (long)TT, SII);
    }

    // 15) shared y = act @ w2s^T
    gemm_wmma<true, false, false><<<dim3(HH / BN, TT / BM, 1), 256, 0, stream>>>(
        act_sh, w2s, sh_y, nullptr, TT, HH, SII, 0, 0, 0);

    // 16) out = x1 + sh_y + moe contributions
    final_combine<<<TT, 256, 0, stream>>>(x1, sh_y, y_moe, flat_e, pos, keep, flat_w, out);
}
